// MultiHeadSelfAttention_62362925138607
// MI455X (gfx1250) — compile-verified, matrix operand order checked by reading, not run on hardware
//
#include <hip/hip_runtime.h>
#include <hip/hip_bf16.h>

#define DM   2048
#define NH   16
#define DH   128
#define TSEQ 2048
#define BB   2

typedef __bf16 bf16_t;
typedef __attribute__((ext_vector_type(16))) __bf16 v16bf;
typedef __attribute__((ext_vector_type(8)))  float  v8f;
typedef __attribute__((ext_vector_type(4)))  int    i32x4;

#define LOG2E 1.4426950408889634f
#define NEGBIG (-3.0e38f)

#if __has_builtin(__builtin_amdgcn_global_load_async_to_lds_b128)
#define HAVE_ASYNC_LDS 1
#else
#define HAVE_ASYNC_LDS 0
#endif

__device__ __forceinline__ void wait_async_zero() {
#if HAVE_ASYNC_LDS
#if __has_builtin(__builtin_amdgcn_s_wait_asynccnt)
    __builtin_amdgcn_s_wait_asynccnt(0);
#else
    asm volatile("s_wait_asynccnt 0x0" ::: "memory");
#endif
#endif
}

__device__ __forceinline__ bf16_t f2bf(float f) {
    unsigned u = __float_as_uint(f);
    unsigned r = (u + 0x7fffu + ((u >> 16) & 1u)) >> 16;
    unsigned short h = (unsigned short)r;
    return __builtin_bit_cast(bf16_t, h);
}

// Load a 16-bit WMMA A/B fragment: 8 elems at p[0..7], 8 at p[16..23]
__device__ __forceinline__ v16bf ld_frag(const bf16_t* p) {
    v16bf f;
    ((uint4*)&f)[0] = *(const uint4*)(p);
    ((uint4*)&f)[1] = *(const uint4*)(p + 16);
    return f;
}

__device__ __forceinline__ v8f wmma_bf16(v16bf a, v16bf b, v8f c) {
    return __builtin_amdgcn_wmma_f32_16x16x32_bf16(false, a, false, b,
                                                   (short)0, c, false, false);
}

// Copy 16 bytes global -> LDS (async path when available)
__device__ __forceinline__ void cp_g2s_b128(bf16_t* dst, const bf16_t* src) {
#if HAVE_ASYNC_LDS
    __builtin_amdgcn_global_load_async_to_lds_b128(
        (i32x4*)src, (i32x4*)dst, 0, 0);
#else
    *(uint4*)dst = *(const uint4*)src;
#endif
}

// ---------------------------------------------------------------- convert
__global__ __launch_bounds__(256) void cvt_kernel(const float* __restrict__ s,
                                                  bf16_t* __restrict__ d, int n) {
    for (int i = blockIdx.x * blockDim.x + threadIdx.x; i < n;
         i += gridDim.x * blockDim.x)
        d[i] = f2bf(s[i]);
}

// ------------------------------------------------- QKV GEMM + RoPE epilogue
// C[m,n] = sum_k X[m,k] * Wqkv[n,k];  M=4096, N=6144, K=2048
// Block tile 64x256 (grid 64x24), wave tile 32x64 (2x4 WMMA accs).
__global__ __launch_bounds__(256) void qkv_rope_kernel(
        const bf16_t* __restrict__ X, const bf16_t* __restrict__ W,
        bf16_t* __restrict__ Qb, bf16_t* __restrict__ Kb, bf16_t* __restrict__ Vb) {
    const int lane = threadIdx.x & 31;
    const int w    = threadIdx.x >> 5;
    const int row0 = blockIdx.x * 64  + (w & 1) * 32;
    const int col0 = blockIdx.y * 256 + (w >> 1) * 64;
    const int l15  = lane & 15;
    const int koff = (lane >> 4) << 3;   // 0 or 8
    const int hi8  = koff;               // row offset for C layout

    const v8f vzero = {0.f,0.f,0.f,0.f,0.f,0.f,0.f,0.f};
    v8f acc[8] = {vzero,vzero,vzero,vzero,vzero,vzero,vzero,vzero};

    const bf16_t* a0 = X + (size_t)(row0 + l15) * DM + koff;
    const bf16_t* b0 = W + (size_t)(col0 + l15) * DM + koff;
    __builtin_prefetch(a0, 0, 1);
    __builtin_prefetch(a0 + (size_t)16 * DM, 0, 1);
#pragma unroll
    for (int j = 0; j < 4; ++j)
        __builtin_prefetch(b0 + (size_t)(j * 16) * DM, 0, 1);

    for (int kb = 0; kb < DM; kb += 32) {
        v16bf av[2], bv[4];
        av[0] = ld_frag(a0 + kb);
        av[1] = ld_frag(a0 + (size_t)16 * DM + kb);
#pragma unroll
        for (int j = 0; j < 4; ++j)
            bv[j] = ld_frag(b0 + (size_t)(j * 16) * DM + kb);
#pragma unroll
        for (int tm = 0; tm < 2; ++tm)
#pragma unroll
            for (int tn = 0; tn < 4; ++tn)
                acc[tm * 4 + tn] = wmma_bf16(av[tm], bv[tn], acc[tm * 4 + tn]);
    }

    for (int tt = 0; tt < 8; ++tt) {
        const int mbase = row0 + ((tt >> 2) << 4);
        const int nbase = col0 + ((tt & 3) << 4);
        for (int i = 0; i < 8; ++i) {
            float v   = acc[tt][i];
            float prt = __shfl_xor(v, 1, 32);       // partner d^1 (adjacent lane)
            const int m = mbase + i + hi8;
            const int n = nbase + l15;
            const int b = m >> 11, tpos = m & 2047;
            const int which = n >> 11;              // 0=q 1=k 2=v
            const int c = n & 2047, h = c >> 7, d = c & 127;
            if (which == 2) {
                // V stored transposed: [B,H,D,T]
                Vb[(((size_t)(b * NH + h)) * DH + d) * TSEQ + tpos] = f2bf(v);
            } else {
                // RoPE: freq = 10000^(-(d&~1)/128)
                float freq = __expf(-0.07195578415606394f * (float)(d & ~1));
                float ang  = (float)tpos * freq;
                float sv, cv;
                __sincosf(ang, &sv, &cv);
                float r = (d & 1) ? (v * cv + prt * sv) : (v * cv - prt * sv);
                bf16_t* dst = (which == 0) ? Qb : Kb;
                dst[(((size_t)(b * NH + h)) * TSEQ + tpos) * DH + d] = f2bf(r);
            }
        }
    }
}

// ---------------------------------------------------------- flash attention
// grid (B*H=32, T/128=16), block 256 (8 waves x 16 q-rows).
__global__ __launch_bounds__(256) void flash_kernel(
        const bf16_t* __restrict__ Qb, const bf16_t* __restrict__ Kb,
        const bf16_t* __restrict__ Vb, bf16_t* __restrict__ Att) {
    __shared__ __align__(16) bf16_t Kt[32 * 136];   // K tile, padded stride
    __shared__ __align__(16) bf16_t Vt[128 * 40];   // V^T tile, padded stride
    __shared__ __align__(16) bf16_t Pt[8 * 16 * 40];// per-wave P tile

    const int tid  = threadIdx.x;
    const int lane = tid & 31;
    const int w    = tid >> 5;
    const int l15  = lane & 15;
    const int koff = (lane >> 4) << 3;
    const int hi8  = koff;
    const int bh   = blockIdx.x;
    const int qt   = blockIdx.y;
    const int qrow0 = qt * 128 + w * 16;

    // Preload Q fragments (16 rows x 128 d = 4 k-chunks)
    v16bf qf[4];
    {
        const bf16_t* qp = Qb + ((size_t)bh * TSEQ + qrow0 + l15) * DH + koff;
        for (int c = 0; c < 4; ++c) qf[c] = ld_frag(qp + c * 32);
    }

    const v8f vzero = {0.f,0.f,0.f,0.f,0.f,0.f,0.f,0.f};
    v8f o[8] = {vzero,vzero,vzero,vzero,vzero,vzero,vzero,vzero};
    float mrow[8], lrow[8];
    for (int i = 0; i < 8; ++i) { mrow[i] = NEGBIG; lrow[i] = 0.f; }

    bf16_t* Pw = &Pt[w * 640];
    const int jmax = qt * 4 + 3;

    for (int j = 0; j <= jmax; ++j) {
        const int kb = j * 32;
        __syncthreads();
        // K tile: 32 keys x 128 d (row-major, stride 136)
        {
            const int ch = tid, ch2 = tid + 256;
            const int r  = ch  >> 4, c16  = (ch  & 15) << 3;
            const int r2 = ch2 >> 4, c16b = (ch2 & 15) << 3;
            cp_g2s_b128(&Kt[r  * 136 + c16 ],
                        Kb + ((size_t)bh * TSEQ + kb + r ) * DH + c16 );
            cp_g2s_b128(&Kt[r2 * 136 + c16b],
                        Kb + ((size_t)bh * TSEQ + kb + r2) * DH + c16b);
        }
        // V^T tile: 128 d x 32 keys (stride 40)
        {
            const int ch = tid, ch2 = tid + 256;
            const int d  = ch  >> 2, c4  = (ch  & 3) << 3;
            const int d2 = ch2 >> 2, c4b = (ch2 & 3) << 3;
            cp_g2s_b128(&Vt[d  * 40 + c4 ],
                        Vb + ((size_t)bh * DH + d ) * TSEQ + kb + c4 );
            cp_g2s_b128(&Vt[d2 * 40 + c4b],
                        Vb + ((size_t)bh * DH + d2) * TSEQ + kb + c4b);
        }
        // Prefetch next tile's source lines while we compute
        if (j < jmax) {
            __builtin_prefetch(Kb + ((size_t)bh * TSEQ + kb + 32 + (tid >> 4)) * DH, 0, 1);
            __builtin_prefetch(Vb + ((size_t)bh * DH + (tid >> 1)) * TSEQ + kb + 32, 0, 1);
        }
        wait_async_zero();
        __syncthreads();

        if (kb <= qrow0 + 15) {   // wave-uniform causal guard
            v8f s0 = vzero, s1 = vzero;
            for (int c = 0; c < 4; ++c) {
                const int base = l15 * 136 + c * 32 + koff;
                v16bf k0 = ld_frag(&Kt[base]);
                v16bf k1 = ld_frag(&Kt[base + 16 * 136]);
                s0 = wmma_bf16(qf[c], k0, s0);
                s1 = wmma_bf16(qf[c], k1, s1);
            }
            const float SC = 0.08838834764831845f;  // 1/sqrt(128)
            for (int i = 0; i < 8; ++i) {
                const int q_idx = qrow0 + i + hi8;
                float a  = s0[i] * SC;
                float b2 = s1[i] * SC;
                if (kb + l15 > q_idx)      a  = NEGBIG;
                if (kb + 16 + l15 > q_idx) b2 = NEGBIG;
                float rm = fmaxf(a, b2);
                for (int msk = 1; msk < 16; msk <<= 1)
                    rm = fmaxf(rm, __shfl_xor(rm, msk, 32));
                float mn   = fmaxf(mrow[i], rm);
                float corr = exp2f((mrow[i] - mn) * LOG2E);
                float p0   = exp2f((a  - mn) * LOG2E);
                float p1   = exp2f((b2 - mn) * LOG2E);
                float rs   = p0 + p1;
                for (int msk = 1; msk < 16; msk <<= 1)
                    rs += __shfl_xor(rs, msk, 32);
                lrow[i] = lrow[i] * corr + rs;
                mrow[i] = mn;
                Pw[(i + hi8) * 40 + l15]      = f2bf(p0);
                Pw[(i + hi8) * 40 + 16 + l15] = f2bf(p1);
                for (int t = 0; t < 8; ++t) o[t][i] *= corr;
            }
            __builtin_amdgcn_wave_barrier();   // order P stores vs reshaped loads
            v16bf pf = ld_frag(&Pw[l15 * 40 + koff]);
            for (int t = 0; t < 8; ++t) {
                v16bf vf = ld_frag(&Vt[(t * 16 + l15) * 40 + koff]);
                o[t] = wmma_bf16(pf, vf, o[t]);
            }
        }
    }

    const int b = bh >> 4, h = bh & 15;
    for (int i = 0; i < 8; ++i) {
        const float rl = 1.0f / lrow[i];
        const int q_idx = qrow0 + i + hi8;
        for (int t = 0; t < 8; ++t)
            Att[((size_t)b * TSEQ + q_idx) * DM + h * DH + t * 16 + l15] =
                f2bf(o[t][i] * rl);
    }
}

// ------------------------------------------------------- output projection
// Out[m,n] = sum_k Att[m,k] * WO[n,k];  M=4096, N=2048, K=2048
// Block tile 64x256 (grid 64x8), wave tile 32x64.
__global__ __launch_bounds__(256) void out_proj_kernel(
        const bf16_t* __restrict__ A, const bf16_t* __restrict__ W,
        float* __restrict__ Out) {
    const int lane = threadIdx.x & 31;
    const int w    = threadIdx.x >> 5;
    const int row0 = blockIdx.x * 64  + (w & 1) * 32;
    const int col0 = blockIdx.y * 256 + (w >> 1) * 64;
    const int l15  = lane & 15;
    const int koff = (lane >> 4) << 3;
    const int hi8  = koff;

    const v8f vzero = {0.f,0.f,0.f,0.f,0.f,0.f,0.f,0.f};
    v8f acc[8] = {vzero,vzero,vzero,vzero,vzero,vzero,vzero,vzero};

    const bf16_t* a0 = A + (size_t)(row0 + l15) * DM + koff;
    const bf16_t* b0 = W + (size_t)(col0 + l15) * DM + koff;
    __builtin_prefetch(a0, 0, 1);
    __builtin_prefetch(a0 + (size_t)16 * DM, 0, 1);
#pragma unroll
    for (int j = 0; j < 4; ++j)
        __builtin_prefetch(b0 + (size_t)(j * 16) * DM, 0, 1);

    for (int kb = 0; kb < DM; kb += 32) {
        v16bf av[2], bv[4];
        av[0] = ld_frag(a0 + kb);
        av[1] = ld_frag(a0 + (size_t)16 * DM + kb);
#pragma unroll
        for (int j = 0; j < 4; ++j)
            bv[j] = ld_frag(b0 + (size_t)(j * 16) * DM + kb);
#pragma unroll
        for (int tm = 0; tm < 2; ++tm)
#pragma unroll
            for (int tn = 0; tn < 4; ++tn)
                acc[tm * 4 + tn] = wmma_bf16(av[tm], bv[tn], acc[tm * 4 + tn]);
    }
    for (int tt = 0; tt < 8; ++tt) {
        const int mbase = row0 + ((tt >> 2) << 4);
        const int nbase = col0 + ((tt & 3) << 4);
        for (int i = 0; i < 8; ++i) {
            const int m = mbase + i + hi8;
            const int n = nbase + l15;
            Out[(size_t)m * DM + n] = acc[tt][i];
        }
    }
}

// ---------------------------------------------------------------- launcher
extern "C" void kernel_launch(void* const* d_in, const int* in_sizes, int n_in,
                              void* d_out, int out_size, void* d_ws, size_t ws_size,
                              hipStream_t stream) {
    (void)in_sizes; (void)n_in; (void)out_size; (void)ws_size;
    const float* x    = (const float*)d_in[0];
    const float* wqkv = (const float*)d_in[1];
    const float* wo   = (const float*)d_in[2];
    float* out = (float*)d_out;

    char* ws = (char*)d_ws;
    bf16_t* xb  = (bf16_t*)(ws);                         // 4096*2048*2  = 16 MB
    bf16_t* wqb = (bf16_t*)(ws + 16777216);              // 6144*2048*2  = 24 MB
    bf16_t* wob = (bf16_t*)(ws + 41943040);              // 2048*2048*2  =  8 MB
    bf16_t* Qb  = (bf16_t*)(ws + 50331648);              // 16 MB
    bf16_t* Kb  = (bf16_t*)(ws + 67108864);              // 16 MB
    bf16_t* Vb  = (bf16_t*)(ws + 83886080);              // 16 MB (transposed)
    bf16_t* Ab  = (bf16_t*)(ws + 100663296);             // 16 MB

    cvt_kernel<<<4096, 256, 0, stream>>>(x,    xb,  BB * TSEQ * DM);
    cvt_kernel<<<4096, 256, 0, stream>>>(wqkv, wqb, 3 * DM * DM);
    cvt_kernel<<<4096, 256, 0, stream>>>(wo,   wob, DM * DM);

    qkv_rope_kernel<<<dim3(64, 24), 256, 0, stream>>>(xb, wqb, Qb, Kb, Vb);
    flash_kernel   <<<dim3(32, 16), 256, 0, stream>>>(Qb, Kb, Vb, Ab);
    out_proj_kernel<<<dim3(64,  8), 256, 0, stream>>>(Ab, wob, out);
}
